// CrossInteractionModule_16458314678571
// MI455X (gfx1250) — matrix operand order repaired and checked by reading, not compiled, not run on hardware
//
#include <hip/hip_runtime.h>
#include <hip/hip_bf16.h>

typedef __attribute__((ext_vector_type(16))) _Float16 v16h;
typedef __attribute__((ext_vector_type(8)))  float    v8f;

#define HD 128          // hidden dim
#define ED 16           // edge attr dim
#define LN_EPS 1e-5f

#define WPB 2           // waves per block (edge kernel)
#define MT  4           // 16-edge M-tiles per wave (edge kernel)

// ---- WMMA fragment index helpers (per CDNA5 ISA 7.12.2) ----
// A-matrix 16x32 f16: element held by (lane, halfslot j) has
//   m = lane & 15,  k(in tile) = a_frag_k(lane, j)
__device__ __forceinline__ int a_frag_k(int lane, int j) {
    int vg = j >> 1, bit = j & 1;
    return ((vg & 4) ? 16 : 0) + ((lane >> 4) << 3) + ((vg & 3) << 1) + bit;
}

__device__ __forceinline__ v8f wmma16(v16h a, v16h b, v8f c) {
    return __builtin_amdgcn_wmma_f32_16x16x32_f16(false, a, false, b, (short)0, c,
                                                  false, false);
}

__device__ __forceinline__ float silu_f(float x) {
    return x / (1.0f + __expf(-x));
}

// Given C-layout element (m, outfeature k in 0..127), compute A-frag LDS slot
// (kt2, lane2, j2) for use as A of the next GEMM.
__device__ __forceinline__ void c_to_afrag(int m, int k, int& kt2, int& lane2, int& j2) {
    kt2 = k >> 5;
    int k32 = k & 31;
    int hi  = (k32 >= 16) ? 1 : 0;
    int rm  = k32 & 15;
    int h8  = rm >> 3;
    int rr  = rm & 7;
    int vg  = (rr >> 1) + (hi ? 4 : 0);
    int bit = rr & 1;
    lane2 = (h8 << 4) + m;
    j2    = (vg << 1) + bit;
}

// ---- weight fp32 [Kreal,128] -> f16 B-fragment layout [KT][8][32][16] ----
__global__ void wfrag_convert(const float* __restrict__ W, _Float16* __restrict__ dst,
                              int Kreal, int KT) {
    int idx = blockIdx.x * blockDim.x + threadIdx.x;
    int total = KT * 8 * 32 * 16;
    if (idx >= total) return;
    int j    = idx & 15;
    int lane = (idx >> 4) & 31;
    int nt   = (idx >> 9) & 7;
    int kt   = idx >> 12;
    int k = kt * 32 + ((j >> 3) << 4) + ((lane >> 4) << 3) + (j & 7);   // B layout: k = 8h+j (j<8), 16+8h+(j-8) (j>=8)
    int n = nt * 16 + (lane & 15);
    float v = (k < Kreal) ? W[k * HD + n] : 0.0f;
    dst[idx] = (_Float16)v;
}

// ---- edge message kernel ----
// One wave owns MT=4 tiles of 16 edges (64 edges). Each B (weight) fragment is
// loaded once from L2 and reused across the 4 M-tiles (4 independent
// accumulators -> 4 back-to-back independent v_wmma per B load).
// Dynamic LDS: A frags [WPB][MT][9][32][16] + hidden frags [WPB][MT][4][32][16].
__global__ __launch_bounds__(WPB * 32) void edge_msg_kernel(
    const float* __restrict__ prot, const float* __restrict__ lig,
    const int* __restrict__ pid, const int* __restrict__ lid,
    const float* __restrict__ attr,
    const _Float16* __restrict__ W1pl, const float* __restrict__ b1pl,
    const _Float16* __restrict__ W2pl, const float* __restrict__ b2pl,
    const _Float16* __restrict__ W1lp, const float* __restrict__ b1lp,
    const _Float16* __restrict__ W2lp, const float* __restrict__ b2lp,
    float* __restrict__ agg_p, float* __restrict__ agg_l, int Ec)
{
    extern __shared__ __align__(32) _Float16 smem[];
    _Float16* Alds = smem;                                // WPB*MT*9*512 halves
    _Float16* Hlds = smem + (size_t)WPB * MT * 9 * 512;   // WPB*MT*4*512 halves

    const int wave = threadIdx.x >> 5;
    const int lane = threadIdx.x & 31;
    const int wbase = (blockIdx.x * WPB + wave) * (MT * 16);

    // ---- Stage gathered edge inputs [p(128)|l(128)|attr(16)|pad] as A frags ----
    #pragma unroll
    for (int mt = 0; mt < MT; ++mt) {
        const int e = wbase + mt * 16 + (lane & 15);
        const bool ev = (e < Ec);
        const int pe = ev ? pid[e] : 0;
        const int le = ev ? lid[e] : 0;
        for (int kt = 0; kt < 9; ++kt) {
            v16h av;
            #pragma unroll
            for (int j = 0; j < 16; ++j) {
                int k = kt * 32 + a_frag_k(lane, j);
                float v = 0.0f;
                if (ev) {
                    if (k < HD)            v = prot[(size_t)pe * HD + k];
                    else if (k < 2 * HD)   v = lig[(size_t)le * HD + (k - HD)];
                    else if (k < 2*HD+ED)  v = attr[(size_t)e * ED + (k - 2 * HD)];
                }
                av[j] = (_Float16)v;
            }
            *(v16h*)&Alds[(size_t)(((wave * MT + mt) * 9 + kt) * 32 + lane) * 16] = av;
        }
    }

    for (int dir = 0; dir < 2; ++dir) {
        const _Float16* W1 = dir ? W1lp : W1pl;
        const float*    b1 = dir ? b1lp : b1pl;
        const _Float16* W2 = dir ? W2lp : W2pl;
        const float*    b2 = dir ? b2lp : b2pl;

        // GEMM1: [64,288] @ [288,128] + b1, SiLU, restage as A frags in Hlds.
        for (int nt = 0; nt < 8; ++nt) {
            float bias = b1[nt * 16 + (lane & 15)];
            v8f acc[MT];
            #pragma unroll
            for (int mt = 0; mt < MT; ++mt)
                #pragma unroll
                for (int r = 0; r < 8; ++r) acc[mt][r] = bias;

            #pragma unroll
            for (int kt = 0; kt < 9; ++kt) {
                // lp input = [l, p, attr]: reuse pl-ordered A frags by K-block swap
                int ks = dir ? ((kt < 4) ? kt + 4 : (kt < 8) ? kt - 4 : 8) : kt;
                v16h b = *(const v16h*)&W1[(size_t)((kt * 8 + nt) * 32 + lane) * 16];
                #pragma unroll
                for (int mt = 0; mt < MT; ++mt) {
                    v16h a = *(const v16h*)
                        &Alds[(size_t)(((wave * MT + mt) * 9 + ks) * 32 + lane) * 16];
                    acc[mt] = wmma16(a, b, acc[mt]);
                }
            }
            #pragma unroll
            for (int mt = 0; mt < MT; ++mt) {
                #pragma unroll
                for (int r = 0; r < 8; ++r) {
                    float s = silu_f(acc[mt][r]);
                    int M = r + ((lane >> 4) << 3);
                    int k = nt * 16 + (lane & 15);
                    int kt2, lane2, j2;
                    c_to_afrag(M, k, kt2, lane2, j2);
                    Hlds[(size_t)(((wave * MT + mt) * 4 + kt2) * 32 + lane2) * 16 + j2] =
                        (_Float16)s;
                }
            }
        }

        // GEMM2: [64,128] @ [128,128] + b2, scatter-add to node aggregates.
        const int* idx = dir ? lid : pid;
        float* agg = dir ? agg_l : agg_p;
        for (int nt = 0; nt < 8; ++nt) {
            float bias = b2[nt * 16 + (lane & 15)];
            v8f acc[MT];
            #pragma unroll
            for (int mt = 0; mt < MT; ++mt)
                #pragma unroll
                for (int r = 0; r < 8; ++r) acc[mt][r] = bias;

            #pragma unroll
            for (int kt = 0; kt < 4; ++kt) {
                v16h b = *(const v16h*)&W2[(size_t)((kt * 8 + nt) * 32 + lane) * 16];
                #pragma unroll
                for (int mt = 0; mt < MT; ++mt) {
                    v16h a = *(const v16h*)
                        &Hlds[(size_t)(((wave * MT + mt) * 4 + kt) * 32 + lane) * 16];
                    acc[mt] = wmma16(a, b, acc[mt]);
                }
            }
            #pragma unroll
            for (int mt = 0; mt < MT; ++mt) {
                #pragma unroll
                for (int r = 0; r < 8; ++r) {
                    int M = r + ((lane >> 4) << 3);
                    int e2 = wbase + mt * 16 + M;
                    if (e2 < Ec) {
                        int node = idx[e2];
                        atomicAdd(&agg[(size_t)node * HD + nt * 16 + (lane & 15)],
                                  acc[mt][r]);
                    }
                }
            }
        }
    }
}

// ---- node update: dp = mlp(concat(h, agg)); h = LN(h + dp) ----
__global__ __launch_bounds__(64) void node_update_kernel(
    float* __restrict__ h, const float* __restrict__ agg,
    const _Float16* __restrict__ W1, const float* __restrict__ b1,
    const _Float16* __restrict__ W2, const float* __restrict__ b2,
    const float* __restrict__ gamma, const float* __restrict__ beta, int n)
{
    __shared__ __align__(32) _Float16 Alds[2][8][32][16];  // K = 256
    __shared__ __align__(32) _Float16 Hlds[2][4][32][16];  // K = 128
    __shared__ float Ylds[2][16][HD];

    const int wave = threadIdx.x >> 5;
    const int lane = threadIdx.x & 31;
    const int tile = blockIdx.x * 2 + wave;
    const int base = tile * 16;
    const int m = lane & 15;
    const int node = base + m;
    const bool nv = (node < n);

    for (int kt = 0; kt < 8; ++kt) {
        v16h av;
        #pragma unroll
        for (int j = 0; j < 16; ++j) {
            int k = kt * 32 + a_frag_k(lane, j);
            float v = 0.0f;
            if (nv) v = (k < HD) ? h[(size_t)node * HD + k]
                                 : agg[(size_t)node * HD + (k - HD)];
            av[j] = (_Float16)v;
        }
        *(v16h*)&Alds[wave][kt][lane][0] = av;
    }

    for (int nt = 0; nt < 8; ++nt) {
        float bias = b1[nt * 16 + (lane & 15)];
        v8f acc;
        #pragma unroll
        for (int r = 0; r < 8; ++r) acc[r] = bias;
        #pragma unroll
        for (int kt = 0; kt < 8; ++kt) {
            v16h a = *(const v16h*)&Alds[wave][kt][lane][0];
            v16h b = *(const v16h*)&W1[(size_t)((kt * 8 + nt) * 32 + lane) * 16];
            acc = wmma16(a, b, acc);
        }
        #pragma unroll
        for (int r = 0; r < 8; ++r) {
            float s = silu_f(acc[r]);
            int M = r + ((lane >> 4) << 3);
            int k = nt * 16 + (lane & 15);
            int kt2, lane2, j2;
            c_to_afrag(M, k, kt2, lane2, j2);
            Hlds[wave][kt2][lane2][j2] = (_Float16)s;
        }
    }

    for (int nt = 0; nt < 8; ++nt) {
        float bias = b2[nt * 16 + (lane & 15)];
        v8f acc;
        #pragma unroll
        for (int r = 0; r < 8; ++r) acc[r] = bias;
        #pragma unroll
        for (int kt = 0; kt < 4; ++kt) {
            v16h a = *(const v16h*)&Hlds[wave][kt][lane][0];
            v16h b = *(const v16h*)&W2[(size_t)((kt * 8 + nt) * 32 + lane) * 16];
            acc = wmma16(a, b, acc);
        }
        #pragma unroll
        for (int r = 0; r < 8; ++r) {
            int M = r + ((lane >> 4) << 3);
            int nd = base + M;
            int c = nt * 16 + (lane & 15);
            if (nd < n) Ylds[wave][M][c] = h[(size_t)nd * HD + c] + acc[r];
        }
    }

    // LayerNorm: lanes 0..15 each own one row (intra-wave DS ordering suffices).
    if (lane < 16) {
        int nd = base + lane;
        if (nd < n) {
            float mu = 0.0f, s2 = 0.0f;
            #pragma unroll 4
            for (int c = 0; c < HD; ++c) {
                float y = Ylds[wave][lane][c];
                mu += y; s2 += y * y;
            }
            mu *= (1.0f / HD);
            float var = s2 * (1.0f / HD) - mu * mu;
            float inv = rsqrtf(var + LN_EPS);
            #pragma unroll 4
            for (int c = 0; c < HD; ++c) {
                float y = Ylds[wave][lane][c];
                h[(size_t)nd * HD + c] = (y - mu) * inv * gamma[c] + beta[c];
            }
        }
    }
}

// fragment element counts
#define F1  (9 * 8 * 32 * 16)   // K=288 (272 padded)
#define F2  (4 * 8 * 32 * 16)   // K=128
#define FU  (8 * 8 * 32 * 16)   // K=256

extern "C" void kernel_launch(void* const* d_in, const int* in_sizes, int n_in,
                              void* d_out, int out_size, void* d_ws, size_t ws_size,
                              hipStream_t stream)
{
    const float* prot_in = (const float*)d_in[0];
    const float* lig_in  = (const float*)d_in[1];
    const int*   cidx    = (const int*)d_in[2];
    const float* attr    = (const float*)d_in[3];
    const int NPn = in_sizes[0] / HD;
    const int NLn = in_sizes[1] / HD;
    const int Ec  = in_sizes[2] / 2;
    const int* pid = cidx;
    const int* lid = cidx + Ec;

    float* prot = (float*)d_out;
    float* lig  = prot + (size_t)NPn * HD;
    hipMemcpyAsync(prot, prot_in, sizeof(float) * (size_t)NPn * HD,
                   hipMemcpyDeviceToDevice, stream);
    hipMemcpyAsync(lig, lig_in, sizeof(float) * (size_t)NLn * HD,
                   hipMemcpyDeviceToDevice, stream);

    char* ws = (char*)d_ws;
    float* agg_p = (float*)ws; ws += sizeof(float) * (size_t)NPn * HD;
    float* agg_l = (float*)ws; ws += sizeof(float) * (size_t)NLn * HD;
    _Float16* W1pl = (_Float16*)ws; ws += sizeof(_Float16) * F1;
    _Float16* W1lp = (_Float16*)ws; ws += sizeof(_Float16) * F1;
    _Float16* W2pl = (_Float16*)ws; ws += sizeof(_Float16) * F2;
    _Float16* W2lp = (_Float16*)ws; ws += sizeof(_Float16) * F2;
    _Float16* W1up = (_Float16*)ws; ws += sizeof(_Float16) * FU;
    _Float16* W2up = (_Float16*)ws; ws += sizeof(_Float16) * F2;
    _Float16* W1ul = (_Float16*)ws; ws += sizeof(_Float16) * FU;
    _Float16* W2ul = (_Float16*)ws; ws += sizeof(_Float16) * F2;

    const int ptiles = (NPn + 15) / 16;
    const int ltiles = (NLn + 15) / 16;
    const int edges_per_block = WPB * MT * 16;
    const int eblocks = (Ec + edges_per_block - 1) / edges_per_block;
    // dynamic LDS for edge kernel: A frags + hidden frags
    const size_t edge_lds = sizeof(_Float16) *
        ((size_t)WPB * MT * 9 * 512 + (size_t)WPB * MT * 4 * 512);   // ~104 KB

    for (int i = 0; i < 2; ++i) {
        // per-layer weight slices (fp32)
        const float* mplW1 = (const float*)d_in[4]  + (size_t)i * 272 * HD;
        const float* mplb1 = (const float*)d_in[5]  + (size_t)i * HD;
        const float* mplW2 = (const float*)d_in[6]  + (size_t)i * HD * HD;
        const float* mplb2 = (const float*)d_in[7]  + (size_t)i * HD;
        const float* mlpW1 = (const float*)d_in[8]  + (size_t)i * 272 * HD;
        const float* mlpb1 = (const float*)d_in[9]  + (size_t)i * HD;
        const float* mlpW2 = (const float*)d_in[10] + (size_t)i * HD * HD;
        const float* mlpb2 = (const float*)d_in[11] + (size_t)i * HD;
        const float* upW1  = (const float*)d_in[12] + (size_t)i * 256 * HD;
        const float* upb1  = (const float*)d_in[13] + (size_t)i * HD;
        const float* upW2  = (const float*)d_in[14] + (size_t)i * HD * HD;
        const float* upb2  = (const float*)d_in[15] + (size_t)i * HD;
        const float* ulW1  = (const float*)d_in[16] + (size_t)i * 256 * HD;
        const float* ulb1  = (const float*)d_in[17] + (size_t)i * HD;
        const float* ulW2  = (const float*)d_in[18] + (size_t)i * HD * HD;
        const float* ulb2  = (const float*)d_in[19] + (size_t)i * HD;
        const float* npg   = (const float*)d_in[20] + (size_t)i * HD;
        const float* npb   = (const float*)d_in[21] + (size_t)i * HD;
        const float* nlg   = (const float*)d_in[22] + (size_t)i * HD;
        const float* nlb   = (const float*)d_in[23] + (size_t)i * HD;

        wfrag_convert<<<(F1 + 255) / 256, 256, 0, stream>>>(mplW1, W1pl, 272, 9);
        wfrag_convert<<<(F1 + 255) / 256, 256, 0, stream>>>(mlpW1, W1lp, 272, 9);
        wfrag_convert<<<(F2 + 255) / 256, 256, 0, stream>>>(mplW2, W2pl, 128, 4);
        wfrag_convert<<<(F2 + 255) / 256, 256, 0, stream>>>(mlpW2, W2lp, 128, 4);
        wfrag_convert<<<(FU + 255) / 256, 256, 0, stream>>>(upW1, W1up, 256, 8);
        wfrag_convert<<<(F2 + 255) / 256, 256, 0, stream>>>(upW2, W2up, 128, 4);
        wfrag_convert<<<(FU + 255) / 256, 256, 0, stream>>>(ulW1, W1ul, 256, 8);
        wfrag_convert<<<(F2 + 255) / 256, 256, 0, stream>>>(ulW2, W2ul, 128, 4);

        hipMemsetAsync(agg_p, 0, sizeof(float) * (size_t)NPn * HD, stream);
        hipMemsetAsync(agg_l, 0, sizeof(float) * (size_t)NLn * HD, stream);

        edge_msg_kernel<<<eblocks, WPB * 32, edge_lds, stream>>>(
            prot, lig, pid, lid, attr,
            W1pl, mplb1, W2pl, mplb2,
            W1lp, mlpb1, W2lp, mlpb2,
            agg_p, agg_l, Ec);

        node_update_kernel<<<(ptiles + 1) / 2, 64, 0, stream>>>(
            prot, agg_p, W1up, upb1, W2up, upb2, npg, npb, NPn);
        node_update_kernel<<<(ltiles + 1) / 2, 64, 0, stream>>>(
            lig, agg_l, W1ul, ulb1, W2ul, ulb2, nlg, nlb, NLn);
    }
}
